// CausalSelfAttention_42949672960534
// MI455X (gfx1250) — matrix operand order repaired and checked by reading, not compiled, not run on hardware
//
#include <hip/hip_runtime.h>
#include <cstdint>
#include <cstddef>

typedef __attribute__((ext_vector_type(16))) __bf16 v16bf;
typedef __attribute__((ext_vector_type(8)))  float  v8f;
typedef unsigned short u16t;

constexpr int Bc = 4, Sc = 2048, Dc = 2048, Hc = 16, HDc = 128;
constexpr int Mrows = Bc * Sc;                       // 8192
constexpr size_t OUT_ELEMS = (size_t)Bc * Sc * Dc;   // 16777216

union Frag16 { v16bf v; uint4 q[2]; u16t u[16]; };
union U4x8   { uint4 q; u16t u[8]; };

__device__ __forceinline__ u16t f2bf(float f) {
  unsigned u = __float_as_uint(f);
  unsigned r = u + 0x7FFFu + ((u >> 16) & 1u);   // round-to-nearest-even
  return (u16t)(r >> 16);
}

// CDNA5 async global->LDS copy (ASYNCcnt-tracked, bypasses VGPRs).
// VDST = 32-bit LDS address (low 32 bits of generic shared pointer),
// VADDR = 64-bit global address (GV mode).
__device__ __forceinline__ void async_copy_b128(void* lds, const void* g) {
  unsigned l = (unsigned)(uintptr_t)lds;
  unsigned long long ga = (unsigned long long)(uintptr_t)g;
  asm volatile("global_load_async_to_lds_b128 %0, %1, off"
               :: "v"(l), "v"(ga) : "memory");
}
__device__ __forceinline__ void wait_async_0() {
  asm volatile("s_wait_asynccnt 0" ::: "memory");
}

// ---------------------------------------------------------------- convert
__global__ void cvt_f32_bf16(const float* __restrict__ in, u16t* __restrict__ out, int n) {
  int i = blockIdx.x * 256 + threadIdx.x;
  if (i < n) out[i] = f2bf(in[i]);
}

// ---------------------------------------------------------------- GEMM  C = A @ W^T
// A: [M x K] bf16 row-major, W: [N x K] bf16 row-major.
// Block tile 64(M) x 64(N); 8 waves: wave (mw,nw) owns two 16x16 C tiles
// (shared A fragment). Double-buffered LDS, one barrier per 32-K step.
// A tile staged with async global->LDS (overlaps WMMAs); W tile register-
// pipelined then scattered into B-fragment layout (lane=n, halves=k).
__global__ void gemm_xwT(const u16t* __restrict__ A, const u16t* __restrict__ W,
                         float* __restrict__ Cf, u16t* __restrict__ Cb,
                         int Nt, int Kt) {
  __shared__ alignas(16) u16t lA[2][4 * 32 * 16];   // [buf][msub][lane][h]
  __shared__ alignas(16) u16t lB[2][4 * 32 * 16];   // [buf][nsub][lane=n][h=k]
  const int t = threadIdx.x, lane = t & 31, wave = t >> 5;
  const int mw = wave >> 1, nw = wave & 1;
  const int M0 = blockIdx.x * 64, N0 = blockIdx.y * 64;
  const int rbase = (lane >> 4) << 3, ncol = lane & 15;

  // A fill: thread -> (row, 8 contiguous k) -> one async b128, swizzled dest
  const int arow = t >> 2, ak0 = (t & 3) * 8;
  const int adst = ((arow >> 4) * 32 + (arow & 15) + ((ak0 >> 3) & 1) * 16) * 16
                   + ((ak0 >> 4) << 3);
  // W fill: thread -> (n, 8 contiguous k) -> uint4 load + 8 scattered b16 stores
  const int bn = t >> 2, bk0 = (t & 3) * 8;
  const int bbase = (bn >> 4) * 512 + ((bn & 15) + ((bk0 >> 3) & 1) * 16) * 16 + ((bk0 >> 4) << 3);

  const u16t* Aptr = A + (size_t)(M0 + arow) * Kt + ak0;
  const u16t* Wptr = W + (size_t)(N0 + bn) * Kt + bk0;

  const v8f vzero = {0.f, 0.f, 0.f, 0.f, 0.f, 0.f, 0.f, 0.f};
  v8f acc0 = vzero, acc1 = vzero;

  // prologue: stage tile 0 into buffer 0
  async_copy_b128(&lA[0][adst], Aptr);
  U4x8 wreg;
  wreg.q = *(const uint4*)(Wptr);
#pragma unroll
  for (int i = 0; i < 8; ++i) lB[0][bbase + i] = wreg.u[i];
  wait_async_0();

  const int T = Kt >> 5;
  int cur = 0;
  for (int it = 0; it < T; ++it) {
    __syncthreads();                       // buf[cur] ready; buf[cur^1] free
    const int nxt = cur ^ 1;
    const bool more = (it + 1 < T);
    U4x8 wnext;
    if (more) {
      async_copy_b128(&lA[nxt][adst], Aptr + (size_t)(it + 1) * 32);
      wnext.q = *(const uint4*)(Wptr + (size_t)(it + 1) * 32);
    }
    Frag16 fa, fb0, fb1;
    const u16t* ap = &lA[cur][(mw * 32 + lane) * 16];
    fa.q[0] = *(const uint4*)(ap);
    fa.q[1] = *(const uint4*)(ap + 8);
    const u16t* bp0 = &lB[cur][((nw * 2 + 0) * 32 + lane) * 16];
    const u16t* bp1 = &lB[cur][((nw * 2 + 1) * 32 + lane) * 16];
    fb0.q[0] = *(const uint4*)bp0;  fb0.q[1] = *(const uint4*)(bp0 + 8);
    fb1.q[0] = *(const uint4*)bp1;  fb1.q[1] = *(const uint4*)(bp1 + 8);
    acc0 = __builtin_amdgcn_wmma_f32_16x16x32_bf16(false, fa.v, false, fb0.v,
                                                   (short)0, acc0, false, false);
    acc1 = __builtin_amdgcn_wmma_f32_16x16x32_bf16(false, fa.v, false, fb1.v,
                                                   (short)0, acc1, false, false);
    if (more) {
#pragma unroll
      for (int i = 0; i < 8; ++i) lB[nxt][bbase + i] = wnext.u[i];
      wait_async_0();                      // tile it+1 async copy complete
    }
    cur = nxt;
  }

  const int row0 = M0 + mw * 16 + rbase;
  const int c0 = N0 + nw * 32 + ncol;
#pragma unroll
  for (int r = 0; r < 8; ++r) {
    size_t rowoff = (size_t)(row0 + r) * Nt;
    if (Cf) { Cf[rowoff + c0] = acc0[r];        Cf[rowoff + c0 + 16] = acc1[r]; }
    if (Cb) { Cb[rowoff + c0] = f2bf(acc0[r]);  Cb[rowoff + c0 + 16] = f2bf(acc1[r]); }
  }
}

// ---------------------------------------------------------------- RoPE + pack to bf16
__global__ void rope_pack(const float* __restrict__ Qf, const float* __restrict__ Kf,
                          const float* __restrict__ cosT, const float* __restrict__ sinT,
                          u16t* __restrict__ Qb, u16t* __restrict__ Kb,
                          float* __restrict__ xk_out) {
  int idx = blockIdx.x * 256 + threadIdx.x;        // over B*S*H*64
  int i = idx & 63;
  int h = (idx >> 6) & 15;
  int s = (idx >> 10) & (Sc - 1);
  int b = idx >> 21;
  size_t base = ((size_t)((b * Sc + s) * Hc + h)) * HDc;
  float c = cosT[s * 64 + i], sn = sinT[s * 64 + i];
  float q1 = Qf[base + i], q2 = Qf[base + 64 + i];
  float k1 = Kf[base + i], k2 = Kf[base + 64 + i];
  float qo1 = q1 * c - q2 * sn, qo2 = q2 * c + q1 * sn;
  float ko1 = k1 * c - k2 * sn, ko2 = k2 * c + k1 * sn;
  Qb[base + i] = f2bf(qo1);      Qb[base + 64 + i] = f2bf(qo2);
  Kb[base + i] = f2bf(ko1);      Kb[base + 64 + i] = f2bf(ko2);
  xk_out[base + i] = ko1;        xk_out[base + 64 + i] = ko2;
}

// ---------------------------------------------------------------- Flash attention
// Grid (S/128, H, B); 256 threads = 8 waves, wave w owns 16 q-rows.
// Double-buffered 32-key K/V tiles, one barrier per tile. V staged via async
// global->LDS; K register-pipelined into B-fragment layout. Online softmax in
// f32; probs through per-wave LDS region to transpose C-layout -> A-layout.
__global__ void flash_fwd(const u16t* __restrict__ Qb, const u16t* __restrict__ Kb,
                          const u16t* __restrict__ Vb, u16t* __restrict__ attnb) {
  __shared__ alignas(16) u16t lK[2][4 * 2 * 32 * 16];   // [buf][kchunk][nhalf][lane=key][h=d]
  __shared__ alignas(16) u16t lV[2][32 * 128];          // [buf][key][d]
  __shared__ alignas(16) u16t lP[8 * 16 * 32];          // per-wave [row][key]
  const int t = threadIdx.x, lane = t & 31, wave = t >> 5;
  const int q0 = blockIdx.x * 128, h = blockIdx.y, b = blockIdx.z;
  const int qw = q0 + wave * 16;
  const int arow = lane & 15, kbase = (lane >> 4) * 8;   // A-frag mapping
  const int rbase = (lane >> 4) << 3;                    // C-layout row base
  const int ncol = lane & 15;                            // C-layout col

  // Q fragments for this wave's 16 rows (HD=128 -> 4 chunks of K=32)
  Frag16 qf[4];
  {
    const u16t* Qrow = Qb + ((size_t)((b * Sc + qw + arow) * Hc + h)) * HDc;
#pragma unroll
    for (int c = 0; c < 4; ++c) {
      qf[c].q[0] = *(const uint4*)(Qrow + c * 32 + kbase);
      qf[c].q[1] = *(const uint4*)(Qrow + c * 32 + 16 + kbase);
    }
  }
  const v8f vzero = {0.f, 0.f, 0.f, 0.f, 0.f, 0.f, 0.f, 0.f};
  v8f O[8];
#pragma unroll
  for (int j = 0; j < 8; ++j) O[j] = vzero;
  float mrow[8], lrow[8];
#pragma unroll
  for (int r = 0; r < 8; ++r) { mrow[r] = -1e30f; lrow[r] = 0.f; }

  // fill mapping: thread -> (key, 16 contiguous d)
  const int fkey = t >> 3, fd0 = (t & 7) * 16;
  const int fs = fkey >> 4, fh = fkey & 15;
  const int kdst = ((fd0 >> 5) * 2 + fs) * 512 + fh * 16; // + (i>>3)*256 + ((fd0>>4)&1)*8 + (i&7)
  const int vdst = fkey * 128 + fd0;
  const size_t tstep = (size_t)32 * Hc * HDc;            // halves per 32-key tile
  const u16t* Kptr = Kb + ((size_t)((b * Sc + fkey) * Hc + h)) * HDc + fd0;
  const u16t* Vptr = Vb + ((size_t)((b * Sc + fkey) * Hc + h)) * HDc + fd0;

  const int ntiles = (q0 + 128) / 32;
  const float scl = 0.088388347648318447f;               // 1/sqrt(128)

  // prologue: stage tile 0 into buffer 0
  async_copy_b128(&lV[0][vdst],     Vptr);
  async_copy_b128(&lV[0][vdst + 8], Vptr + 8);
  Frag16 kreg;
  kreg.q[0] = *(const uint4*)(Kptr);
  kreg.q[1] = *(const uint4*)(Kptr + 8);
#pragma unroll
  for (int i = 0; i < 16; ++i) lK[0][kdst + (i >> 3) * 256 + (((fd0 >> 4) & 1) << 3) + (i & 7)] = kreg.u[i];
  wait_async_0();

  int cur = 0;
  for (int ktile = 0; ktile < ntiles; ++ktile) {
    __syncthreads();                     // buf[cur] ready; buf[cur^1] free
    const int nxt = cur ^ 1;
    const bool more = (ktile + 1 < ntiles);
    const int key0 = ktile * 32;
    Frag16 knext;
    if (more) {
      const u16t* Vn = Vptr + (size_t)(ktile + 1) * tstep;
      async_copy_b128(&lV[nxt][vdst],     Vn);
      async_copy_b128(&lV[nxt][vdst + 8], Vn + 8);
      const u16t* Kn = Kptr + (size_t)(ktile + 1) * tstep;
      knext.q[0] = *(const uint4*)(Kn);
      knext.q[1] = *(const uint4*)(Kn + 8);
    }

    // S = Q K^T : two 16-key halves, K-dim 128 -> 8 WMMAs
    v8f s0 = vzero, s1 = vzero;
#pragma unroll
    for (int c = 0; c < 4; ++c) {
      Frag16 k0f, k1f;
      const u16t* p0 = &lK[cur][((c * 2 + 0) * 32 + lane) * 16];
      const u16t* p1 = &lK[cur][((c * 2 + 1) * 32 + lane) * 16];
      k0f.q[0] = *(const uint4*)p0;  k0f.q[1] = *(const uint4*)(p0 + 8);
      k1f.q[0] = *(const uint4*)p1;  k1f.q[1] = *(const uint4*)(p1 + 8);
      s0 = __builtin_amdgcn_wmma_f32_16x16x32_bf16(false, qf[c].v, false, k0f.v,
                                                   (short)0, s0, false, false);
      s1 = __builtin_amdgcn_wmma_f32_16x16x32_bf16(false, qf[c].v, false, k1f.v,
                                                   (short)0, s1, false, false);
    }

    // causal mask + scale, row max (shfl-xor over the 16-lane half)
    float tmax[8];
#pragma unroll
    for (int r = 0; r < 8; ++r) {
      int qg = qw + rbase + r;
      float a0 = (key0 + ncol      <= qg) ? s0[r] * scl : -1e30f;
      float a1 = (key0 + 16 + ncol <= qg) ? s1[r] * scl : -1e30f;
      s0[r] = a0; s1[r] = a1;
      tmax[r] = fmaxf(a0, a1);
    }
#pragma unroll
    for (int msk = 1; msk < 16; msk <<= 1)
#pragma unroll
      for (int r = 0; r < 8; ++r) tmax[r] = fmaxf(tmax[r], __shfl_xor(tmax[r], msk, 32));

    float alpha[8], rsum[8];
#pragma unroll
    for (int r = 0; r < 8; ++r) {
      float mn = fmaxf(mrow[r], tmax[r]);
      alpha[r] = __expf(mrow[r] - mn);
      mrow[r] = mn;
      float p0 = __expf(s0[r] - mn), p1 = __expf(s1[r] - mn);
      s0[r] = p0; s1[r] = p1;
      rsum[r] = p0 + p1;
    }
#pragma unroll
    for (int msk = 1; msk < 16; msk <<= 1)
#pragma unroll
      for (int r = 0; r < 8; ++r) rsum[r] += __shfl_xor(rsum[r], msk, 32);
#pragma unroll
    for (int r = 0; r < 8; ++r) lrow[r] = lrow[r] * alpha[r] + rsum[r];
#pragma unroll
    for (int j = 0; j < 8; ++j)
#pragma unroll
      for (int r = 0; r < 8; ++r) O[j][r] *= alpha[r];

    // probs: C-layout -> LDS -> A-layout fragment (own-wave region, DS in-order)
    u16t* Pw = lP + wave * 512;
#pragma unroll
    for (int r = 0; r < 8; ++r) {
      Pw[(rbase + r) * 32 + ncol]      = f2bf(s0[r]);
      Pw[(rbase + r) * 32 + ncol + 16] = f2bf(s1[r]);
    }
    Frag16 pf;
    const u16t* Pr = lP + wave * 512 + arow * 32;
    pf.q[0] = *(const uint4*)(Pr + kbase);
    pf.q[1] = *(const uint4*)(Pr + 16 + kbase);

    // O += P @ V : 8 d-subtiles -> 8 WMMAs
#pragma unroll
    for (int j = 0; j < 8; ++j) {
      Frag16 vf;
      const u16t* vp = &lV[cur][(lane >> 4) * 8 * 128 + j * 16 + ncol];
#pragma unroll
      for (int s = 0; s < 8; ++s) { vf.u[s] = vp[s * 128]; vf.u[s + 8] = vp[(16 + s) * 128]; }
      O[j] = __builtin_amdgcn_wmma_f32_16x16x32_bf16(false, pf.v, false, vf.v,
                                                     (short)0, O[j], false, false);
    }

    if (more) {
#pragma unroll
      for (int i = 0; i < 16; ++i) lK[nxt][kdst + (i >> 3) * 256 + (((fd0 >> 4) & 1) << 3) + (i & 7)] = knext.u[i];
      wait_async_0();                    // next V tile async copy complete
    }
    cur = nxt;
  }

  // epilogue: O / l -> bf16 attention output [b*S+s][h*128+d]
#pragma unroll
  for (int j = 0; j < 8; ++j)
#pragma unroll
    for (int r = 0; r < 8; ++r) {
      int row = qw + rbase + r;
      attnb[(size_t)(b * Sc + row) * Dc + h * HDc + j * 16 + ncol] =
          f2bf(O[j][r] / lrow[r]);
    }
}

// ---------------------------------------------------------------- launch
extern "C" void kernel_launch(void* const* d_in, const int* in_sizes, int n_in,
                              void* d_out, int out_size, void* d_ws, size_t ws_size,
                              hipStream_t stream) {
  (void)in_sizes; (void)n_in; (void)out_size; (void)ws_size;
  const float* x  = (const float*)d_in[0];
  const float* wq = (const float*)d_in[1];
  const float* wk = (const float*)d_in[2];
  const float* wv = (const float*)d_in[3];
  const float* wo = (const float*)d_in[4];
  const float* fc = (const float*)d_in[5];
  const float* fs = (const float*)d_in[6];

  float* out    = (float*)d_out;
  float* xk_out = out + OUT_ELEMS;
  float* xv_out = out + 2 * OUT_ELEMS;

  char* ws = (char*)d_ws;
  u16t*  xb  = (u16t*)(ws);                       //  33.5 MB  x bf16
  u16t*  wqb = (u16t*)(ws + 33554432);            //   8.4 MB
  u16t*  wkb = (u16t*)(ws + 41943040);
  u16t*  wvb = (u16t*)(ws + 50331648);
  u16t*  wob = (u16t*)(ws + 58720256);
  float* QF  = (float*)(ws + 67108864);           //  67 MB  Q f32 (pre-RoPE)
  float* KF  = (float*)(ws + 134217728);          //  67 MB  K f32 (pre-RoPE)
  u16t*  QB  = (u16t*)(ws + 201326592);           //  33.5 MB Q bf16 post-RoPE
  u16t*  KB  = (u16t*)(ws + 234881024);           //  33.5 MB K bf16 post-RoPE
  u16t*  VB  = (u16t*)(ws + 268435456);           //  33.5 MB V bf16
  u16t*  AB  = (u16t*)(ws + 67108864);            //  reuse QF region for attn bf16

  const int nx = Bc * Sc * Dc;     // 16777216
  const int nw = Dc * Dc;          //  4194304
  cvt_f32_bf16<<<nx / 256, 256, 0, stream>>>(x,  xb,  nx);
  cvt_f32_bf16<<<nw / 256, 256, 0, stream>>>(wq, wqb, nw);
  cvt_f32_bf16<<<nw / 256, 256, 0, stream>>>(wk, wkb, nw);
  cvt_f32_bf16<<<nw / 256, 256, 0, stream>>>(wv, wvb, nw);
  cvt_f32_bf16<<<nw / 256, 256, 0, stream>>>(wo, wob, nw);

  dim3 gg(Mrows / 64, Dc / 64);
  gemm_xwT<<<gg, 256, 0, stream>>>(xb, wqb, QF, nullptr, Dc, Dc);
  gemm_xwT<<<gg, 256, 0, stream>>>(xb, wkb, KF, nullptr, Dc, Dc);
  gemm_xwT<<<gg, 256, 0, stream>>>(xb, wvb, xv_out, VB, Dc, Dc);   // xv output + bf16 V

  const int nr = Bc * Sc * Hc * 64;   // 8388608
  rope_pack<<<nr / 256, 256, 0, stream>>>(QF, KF, fc, fs, QB, KB, xk_out);

  flash_fwd<<<dim3(Sc / 128, Hc, Bc), 256, 0, stream>>>(QB, KB, VB, AB);

  gemm_xwT<<<gg, 256, 0, stream>>>(AB, wob, out, nullptr, Dc, Dc);
}
